// SAMPO_Module_60756607369495
// MI455X (gfx1250) — matrix operand order repaired and checked by reading, not compiled, not run on hardware
//
#include <hip/hip_runtime.h>

typedef _Float16 v16h __attribute__((ext_vector_type(16)));
typedef _Float16 v8h  __attribute__((ext_vector_type(8)));
typedef float    v8f  __attribute__((ext_vector_type(8)));

#define B_TOTAL 4096
#define IN_ROW  4144   // 64*32 + 64*32 + 48
#define SELF_OFF 4096
#define CAT_W   4224
#define OUT_W   132
#define SAMPO_SMEM 160768

// ---------------------------------------------------------------- helpers

__device__ __forceinline__ v8f wmma_f16(v16h a, v16h b, v8f c) {
  // v_wmma_f32_16x16x32_f16  (neg_a, A, neg_b, B, c_mod, C, reuse_a, reuse_b)
  return __builtin_amdgcn_wmma_f32_16x16x32_f16(false, a, false, b, (short)0, c,
                                                false, false);
}

__device__ __forceinline__ v8f zero8() {
  v8f z;
#pragma unroll
  for (int i = 0; i < 8; ++i) z[i] = 0.0f;
  return z;
}

// A fragment (16x32 f16) from a row-major tile. ISA 7.12.2:
// lane L: m = L&15, k = {kb..kb+7, 16+kb..16+kb+7}, kb = (L>=16)?8:0
__device__ __forceinline__ v16h load_a_frag(const _Float16* base, int stride, int lane) {
  int m  = lane & 15;
  int kb = (lane >> 4) << 3;
  const _Float16* p = base + m * stride + kb;
  v8h lo = *(const v8h*)(p);
  v8h hi = *(const v8h*)(p + 16);
  v16h a;
#pragma unroll
  for (int i = 0; i < 8; ++i) { a[i] = lo[i]; a[i + 8] = hi[i]; }
  return a;
}

// B fragment from pre-swizzled weights: block = 512 halves, lane-major.
__device__ __forceinline__ v16h load_b_frag(const _Float16* wsw, int blk, int lane) {
  return *(const v16h*)(wsw + ((size_t)blk << 9) + (lane << 4));
}

// CDNA5 async copy: 16 bytes/lane global -> LDS, tracked by ASYNCcnt.
__device__ __forceinline__ void async_tile_load(unsigned int lds_byte_addr,
                                                unsigned int voff_bytes,
                                                unsigned long long saddr) {
  asm volatile("global_load_async_to_lds_b128 %0, %1, %2"
               :
               : "v"(lds_byte_addr), "v"(voff_bytes), "s"(saddr)
               : "memory");
}
__device__ __forceinline__ void wait_async0() {
  asm volatile("s_wait_asynccnt 0x0" ::: "memory");
}

// ------------------------------------------------ weight swizzle (f32 -> f16)
// dst layout: block (nt*KT + kt), half index = lane*16 + i,
//   k = kt*32 + (lane>=16 ? 8 : 0) + (i&7) + (i>=8 ? 16 : 0),  n = nt*16 + (lane&15)
__global__ __launch_bounds__(256) void k_swizzle(const float* __restrict__ src,
                                                 _Float16* __restrict__ dst,
                                                 int K, int N, int KT, int NT) {
  size_t total = (size_t)KT * NT * 512;
  for (size_t t = (size_t)blockIdx.x * 256 + threadIdx.x; t < total;
       t += (size_t)gridDim.x * 256) {
    int i    = (int)(t & 15);
    int lane = (int)((t >> 4) & 31);
    size_t blk = t >> 9;
    int kt = (int)(blk % KT);
    int nt = (int)(blk / KT);
    int k = kt * 32 + ((lane >= 16) ? 8 : 0) + (i & 7) + ((i >= 8) ? 16 : 0);
    int n = nt * 16 + (lane & 15);
    float v = (k < K && n < N) ? src[(size_t)k * N + n] : 0.0f;
    dst[t] = (_Float16)v;
  }
}

// ------------------------------------------------ entity MLP (ally+enemy)
// rows = B*128 entities, (r,32)@(32,128) relu @(128,128) relu -> ve f16
__global__ __launch_bounds__(256) void k_entity_mlp(
    const float* __restrict__ input,
    const _Float16* __restrict__ W1s, const float* __restrict__ b1,
    const _Float16* __restrict__ W2s, const float* __restrict__ b2,
    _Float16* __restrict__ ve) {
  __shared__ __align__(16) _Float16 sIn[64 * 32];
  __shared__ __align__(16) _Float16 sH[64 * 128];
  const int tid = threadIdx.x, lane = tid & 31, wid = tid >> 5;
  const size_t rbase = (size_t)blockIdx.x * 64;
  for (int idx = tid; idx < 64 * 32; idx += 256) {
    int row = idx >> 5, f = idx & 31;
    size_t r = rbase + row;
    size_t b = r >> 7;
    int e = (int)(r & 127);
    sIn[idx] = (_Float16)input[b * IN_ROW + e * 32 + f];
  }
  __syncthreads();
  const int mt = wid & 3, n0 = (wid >> 2) << 2;
  const int m16 = lane & 15, hi8 = (lane >> 4) << 3;
  {  // GEMM1, K=32 (one wmma per tile)
    v16h a = load_a_frag(sIn + mt * 16 * 32, 32, lane);
#pragma unroll
    for (int j = 0; j < 4; ++j) {
      int nt = n0 + j;
      v8f acc = zero8();
      acc = wmma_f16(a, load_b_frag(W1s, nt, lane), acc);
      float bias = b1[nt * 16 + m16];
#pragma unroll
      for (int i = 0; i < 8; ++i) {
        float v = fmaxf(acc[i] + bias, 0.0f);
        sH[(mt * 16 + i + hi8) * 128 + nt * 16 + m16] = (_Float16)v;
      }
    }
  }
  __syncthreads();
#pragma unroll
  for (int j = 0; j < 4; ++j) {  // GEMM2, K=128
    int nt = n0 + j;
    v8f acc = zero8();
#pragma unroll
    for (int kt = 0; kt < 4; ++kt) {
      v16h a = load_a_frag(sH + mt * 16 * 128 + kt * 32, 128, lane);
      acc = wmma_f16(a, load_b_frag(W2s, nt * 4 + kt, lane), acc);
    }
    float bias = b2[nt * 16 + m16];
#pragma unroll
    for (int i = 0; i < 8; ++i) {
      float v = fmaxf(acc[i] + bias, 0.0f);
      ve[(rbase + mt * 16 + i + hi8) * 128 + nt * 16 + m16] = (_Float16)v;
    }
  }
}

// ------------------------------------------------ self MLP, K padded 48->64
__global__ __launch_bounds__(256) void k_self_mlp(
    const float* __restrict__ input,
    const _Float16* __restrict__ W1s, const float* __restrict__ b1,
    const _Float16* __restrict__ W2s, const float* __restrict__ b2,
    _Float16* __restrict__ vself, _Float16* __restrict__ cat) {
  __shared__ __align__(16) _Float16 sIn[64 * 64];
  __shared__ __align__(16) _Float16 sH[64 * 128];
  const int tid = threadIdx.x, lane = tid & 31, wid = tid >> 5;
  const size_t rbase = (size_t)blockIdx.x * 64;
  for (int idx = tid; idx < 64 * 64; idx += 256) {
    int row = idx >> 6, f = idx & 63;
    float v = (f < 48) ? input[(rbase + row) * IN_ROW + SELF_OFF + f] : 0.0f;
    sIn[idx] = (_Float16)v;
  }
  __syncthreads();
  const int mt = wid & 3, n0 = (wid >> 2) << 2;
  const int m16 = lane & 15, hi8 = (lane >> 4) << 3;
#pragma unroll
  for (int j = 0; j < 4; ++j) {  // GEMM1, K=64
    int nt = n0 + j;
    v8f acc = zero8();
#pragma unroll
    for (int kt = 0; kt < 2; ++kt) {
      v16h a = load_a_frag(sIn + mt * 16 * 64 + kt * 32, 64, lane);
      acc = wmma_f16(a, load_b_frag(W1s, nt * 2 + kt, lane), acc);
    }
    float bias = b1[nt * 16 + m16];
#pragma unroll
    for (int i = 0; i < 8; ++i) {
      float v = fmaxf(acc[i] + bias, 0.0f);
      sH[(mt * 16 + i + hi8) * 128 + nt * 16 + m16] = (_Float16)v;
    }
  }
  __syncthreads();
#pragma unroll
  for (int j = 0; j < 4; ++j) {  // GEMM2, K=128
    int nt = n0 + j;
    v8f acc = zero8();
#pragma unroll
    for (int kt = 0; kt < 4; ++kt) {
      v16h a = load_a_frag(sH + mt * 16 * 128 + kt * 32, 128, lane);
      acc = wmma_f16(a, load_b_frag(W2s, nt * 4 + kt, lane), acc);
    }
    float bias = b2[nt * 16 + m16];
#pragma unroll
    for (int i = 0; i < 8; ++i) {
      float v = fmaxf(acc[i] + bias, 0.0f);
      _Float16 h = (_Float16)v;
      size_t row = rbase + mt * 16 + i + hi8;
      vself[row * 128 + nt * 16 + m16] = h;
      cat[row * CAT_W + 4096 + nt * 16 + m16] = h;  // tail of DS input
    }
  }
}

// ------------------------------------------------ fused SAMPO
// Per block: 16 batches, one side. u=U^T v, Q=Vq^T u, w=Vk Q, t=U w,
// compat = ve . t ; top-16 (softmax monotonic -> skipped) ; gather -> cat
__global__ __launch_bounds__(256) void k_sampo(
    const _Float16* __restrict__ ve_ent, const _Float16* __restrict__ vself,
    const _Float16* __restrict__ Wu_a, const float* __restrict__ bu_a,
    const _Float16* __restrict__ Wvq_a, const float* __restrict__ bvq_a,
    const _Float16* __restrict__ Wvk_a, const float* __restrict__ bvk_a,
    const _Float16* __restrict__ Wu_e, const float* __restrict__ bu_e,
    const _Float16* __restrict__ Wvq_e, const float* __restrict__ bvq_e,
    const _Float16* __restrict__ Wvk_e, const float* __restrict__ bvk_e,
    _Float16* __restrict__ cat) {
  extern __shared__ unsigned char smem[];
  _Float16* sV = (_Float16*)smem;                     // 16x128       4096 B
  _Float16* sU = (_Float16*)(smem + 4096);            // 16x4096    131072 B
  float* su = (float*)(smem + 4096 + 131072);         // 16x32
  float* sQ = su + 512;                               // 16x128
  float* sw = sQ + 2048;                              // 16x32
  float* st = sw + 512;                               // 16x128
  float* sc = st + 2048;                              // 16x64
  int* ssel = (int*)(sc + 1024);                      // 16x16
  const int tid = threadIdx.x, lane = tid & 31, wid = tid >> 5;
  const int side = blockIdx.y;
  const size_t bbase = (size_t)blockIdx.x * 16;
  const _Float16* Wu  = side ? Wu_e  : Wu_a;  const float* bu  = side ? bu_e  : bu_a;
  const _Float16* Wvq = side ? Wvq_e : Wvq_a; const float* bvq = side ? bvq_e : bvq_a;
  const _Float16* Wvk = side ? Wvk_e : Wvk_a; const float* bvk = side ? bvk_e : bvk_a;

  ((uint4*)sV)[tid] = ((const uint4*)(vself + bbase * 128))[tid];  // 256*16B
  for (int i = tid; i < 512 + 2048 + 512; i += 256) su[i] = 0.0f;  // u,Q,w
  __syncthreads();

  const int m16 = lane & 15, hi8 = (lane >> 4) << 3;
  v16h a0 = load_a_frag(sV + 0, 128, lane);
  v16h a1 = load_a_frag(sV + 32, 128, lane);
  v16h a2 = load_a_frag(sV + 64, 128, lane);
  v16h a3 = load_a_frag(sV + 96, 128, lane);

  // Phase 1: U = v@Wu + bu (store f16), u[b,r] += v[b,h]*U  (col = 32h + r)
  for (int nt = wid; nt < 256; nt += 8) {
    v8f acc = zero8();
    acc = wmma_f16(a0, load_b_frag(Wu, nt * 4 + 0, lane), acc);
    acc = wmma_f16(a1, load_b_frag(Wu, nt * 4 + 1, lane), acc);
    acc = wmma_f16(a2, load_b_frag(Wu, nt * 4 + 2, lane), acc);
    acc = wmma_f16(a3, load_b_frag(Wu, nt * 4 + 3, lane), acc);
    int col = nt * 16 + m16;
    float bias = bu[col];
    int h = col >> 5, r = col & 31;
#pragma unroll
    for (int i = 0; i < 8; ++i) {
      int b = i + hi8;
      float v = acc[i] + bias;
      sU[b * 4096 + col] = (_Float16)v;
      atomicAdd(&su[b * 32 + r], (float)sV[b * 128 + h] * v);
    }
  }
  __syncthreads();
  // Phase 2: Q[b,k] += u[b,r] * Vq[b,r,k]   (col = 128r + k)
  for (int nt = wid; nt < 256; nt += 8) {
    v8f acc = zero8();
    acc = wmma_f16(a0, load_b_frag(Wvq, nt * 4 + 0, lane), acc);
    acc = wmma_f16(a1, load_b_frag(Wvq, nt * 4 + 1, lane), acc);
    acc = wmma_f16(a2, load_b_frag(Wvq, nt * 4 + 2, lane), acc);
    acc = wmma_f16(a3, load_b_frag(Wvq, nt * 4 + 3, lane), acc);
    int col = nt * 16 + m16;
    float bias = bvq[col];
    int r = col >> 7, k = col & 127;
#pragma unroll
    for (int i = 0; i < 8; ++i) {
      int b = i + hi8;
      atomicAdd(&sQ[b * 128 + k], su[b * 32 + r] * (acc[i] + bias));
    }
  }
  __syncthreads();
  // Phase 3: w[b,r] += Vk[b,r,k] * Q[b,k]
  for (int nt = wid; nt < 256; nt += 8) {
    v8f acc = zero8();
    acc = wmma_f16(a0, load_b_frag(Wvk, nt * 4 + 0, lane), acc);
    acc = wmma_f16(a1, load_b_frag(Wvk, nt * 4 + 1, lane), acc);
    acc = wmma_f16(a2, load_b_frag(Wvk, nt * 4 + 2, lane), acc);
    acc = wmma_f16(a3, load_b_frag(Wvk, nt * 4 + 3, lane), acc);
    int col = nt * 16 + m16;
    float bias = bvk[col];
    int r = col >> 7, k = col & 127;
#pragma unroll
    for (int i = 0; i < 8; ++i) {
      int b = i + hi8;
      atomicAdd(&sw[b * 32 + r], (acc[i] + bias) * sQ[b * 128 + k]);
    }
  }
  __syncthreads();
  // Phase 4: t[b,h] = (1/sqrt(128)) * sum_r U[b,h,r]*w[b,r]
  for (int o = tid; o < 2048; o += 256) {
    int b = o >> 7, h = o & 127;
    float s = 0.0f;
#pragma unroll
    for (int r = 0; r < 32; ++r) s += (float)sU[b * 4096 + h * 32 + r] * sw[b * 32 + r];
    st[o] = s * 0.08838834764831845f;
  }
  __syncthreads();
  // Phase 5: compat[b,a] = ve[b,a,:] . t[b,:]
  for (int o = tid; o < 1024; o += 256) {
    int b = o >> 6, a = o & 63;
    const _Float16* vp = ve_ent + ((bbase + b) * 128 + side * 64 + a) * 128;
    float s = 0.0f;
    for (int h = 0; h < 128; h += 8) {
      v8h c8 = *(const v8h*)(vp + h);
#pragma unroll
      for (int i = 0; i < 8; ++i) s += (float)c8[i] * st[b * 128 + h + i];
    }
    sc[o] = s;
  }
  __syncthreads();
  // Phase 6: top-16 indices (ties -> lowest index, like lax.top_k)
  if (tid < 16) {
    int b = tid;
    for (int j = 0; j < 16; ++j) {
      float best = -3.0e38f;
      int bi = 0;
      for (int a = 0; a < 64; ++a) {
        float v = sc[b * 64 + a];
        if (v > best) { best = v; bi = a; }
      }
      ssel[b * 16 + j] = bi;
      sc[b * 64 + bi] = -3.4e38f;
    }
  }
  __syncthreads();
  // Phase 7: gather selected rows -> cat (16B chunks)
  for (int cc = tid; cc < 4096; cc += 256) {
    int b = cc >> 8, j = (cc >> 4) & 15, ch = cc & 15;
    int a = ssel[b * 16 + j];
    const uint4* src =
        (const uint4*)(ve_ent + ((bbase + b) * 128 + side * 64 + a) * 128) + ch;
    uint4* dst =
        (uint4*)(cat + (bbase + b) * CAT_W + side * 2048 + j * 128) + ch;
    *dst = *src;
  }
}

// ------------------------------------------------ DS MLP: (B,4224)->(128)->(132)
// K-loop double-buffers cat tiles via GLOBAL_LOAD_ASYNC_TO_LDS_B128 (ASYNCcnt),
// overlapping the staging copy with the WMMA stream; one barrier per iteration.
__global__ __launch_bounds__(256) void k_ds_mlp(
    const _Float16* __restrict__ cat,
    const _Float16* __restrict__ W1s, const float* __restrict__ b1,
    const _Float16* __restrict__ W2s, const float* __restrict__ b2,
    float* __restrict__ out) {
  __shared__ __align__(16) _Float16 sA[2][64 * 32];
  __shared__ __align__(16) _Float16 sH[64 * 128];
  const int tid = threadIdx.x, lane = tid & 31, wid = tid >> 5;
  const size_t rbase = (size_t)blockIdx.x * 64;
  const int mt = wid & 3, n0 = (wid >> 2) << 2;
  const int m16 = lane & 15, hi8 = (lane >> 4) << 3;
  v8f acc[4] = {zero8(), zero8(), zero8(), zero8()};
  const int srow = tid >> 2, scq = tid & 3;
  // lane-private addresses for the async tile copies (16 B per lane)
  const unsigned int lds0 = (unsigned int)(uintptr_t)&sA[0][0] + (unsigned int)tid * 16u;
  const unsigned int lds1 = (unsigned int)(uintptr_t)&sA[1][0] + (unsigned int)tid * 16u;
  const unsigned int voff = (unsigned int)(srow * (CAT_W * 2) + scq * 16);
  const unsigned long long gbase =
      (unsigned long long)(uintptr_t)cat + (unsigned long long)rbase * (CAT_W * 2);
  async_tile_load(lds0, voff, gbase);  // tile 0 -> buf 0
  wait_async0();
  __syncthreads();
  for (int kt = 0; kt < 132; ++kt) {
    const _Float16* cur = sA[kt & 1];
    if (kt + 1 < 132) {
      async_tile_load((kt & 1) ? lds0 : lds1, voff,
                      gbase + (unsigned long long)(kt + 1) * 64u);
      __builtin_prefetch(W1s + (((size_t)n0 * 132 + kt + 1) << 9), 0, 3);
    }
    v16h a = load_a_frag(cur + mt * 16 * 32, 32, lane);
#pragma unroll
    for (int j = 0; j < 4; ++j)
      acc[j] = wmma_f16(a, load_b_frag(W1s, (n0 + j) * 132 + kt, lane), acc[j]);
    wait_async0();   // own async load (tile kt+1) complete
    __syncthreads(); // all waves' loads complete -> next tile visible; WAR closed
  }
#pragma unroll
  for (int j = 0; j < 4; ++j) {
    float bias = b1[(n0 + j) * 16 + m16];
#pragma unroll
    for (int i = 0; i < 8; ++i) {
      float v = fmaxf(acc[j][i] + bias, 0.0f);
      sH[(mt * 16 + i + hi8) * 128 + (n0 + j) * 16 + m16] = (_Float16)v;
    }
  }
  __syncthreads();
  // GEMM2: K=128, N padded to 144 (9 ntiles) -> 36 tiles over 8 waves
  for (int tt = wid; tt < 36; tt += 8) {
    int mt2 = tt & 3, nt = tt >> 2;
    v8f a2 = zero8();
#pragma unroll
    for (int kt = 0; kt < 4; ++kt) {
      v16h a = load_a_frag(sH + mt2 * 16 * 128 + kt * 32, 128, lane);
      a2 = wmma_f16(a, load_b_frag(W2s, nt * 4 + kt, lane), a2);
    }
    int n = nt * 16 + m16;
    if (n < OUT_W) {
      float bias = b2[n];
#pragma unroll
      for (int i = 0; i < 8; ++i) {
        size_t row = rbase + mt2 * 16 + i + hi8;
        out[row * OUT_W + n] = fmaxf(a2[i] + bias, 0.0f);
      }
    }
  }
}

// ------------------------------------------------ host side
extern "C" void kernel_launch(void* const* d_in, const int* in_sizes, int n_in,
                              void* d_out, int out_size, void* d_ws, size_t ws_size,
                              hipStream_t stream) {
  (void)in_sizes; (void)n_in; (void)out_size; (void)ws_size;
  const float* input     = (const float*)d_in[0];
  const float* other_W1  = (const float*)d_in[1];
  const float* other_b1  = (const float*)d_in[2];
  const float* other_W2  = (const float*)d_in[3];
  const float* other_b2  = (const float*)d_in[4];
  const float* self_W1   = (const float*)d_in[5];
  const float* self_b1   = (const float*)d_in[6];
  const float* self_W2   = (const float*)d_in[7];
  const float* self_b2   = (const float*)d_in[8];
  const float* ds_W1     = (const float*)d_in[9];
  const float* ds_b1     = (const float*)d_in[10];
  const float* ds_W2     = (const float*)d_in[11];
  const float* ds_b2     = (const float*)d_in[12];
  const float* ally_Wu   = (const float*)d_in[13];
  const float* ally_bu   = (const float*)d_in[14];
  const float* ally_Wvq  = (const float*)d_in[15];
  const float* ally_bvq  = (const float*)d_in[16];
  const float* ally_Wvk  = (const float*)d_in[17];
  const float* ally_bvk  = (const float*)d_in[18];
  const float* enemy_Wu  = (const float*)d_in[19];
  const float* enemy_bu  = (const float*)d_in[20];
  const float* enemy_Wvq = (const float*)d_in[21];
  const float* enemy_bvq = (const float*)d_in[22];
  const float* enemy_Wvk = (const float*)d_in[23];
  const float* enemy_bvk = (const float*)d_in[24];
  float* out = (float*)d_out;

  unsigned char* p = (unsigned char*)d_ws;
  auto carve = [&](size_t halves) -> _Float16* {
    _Float16* r = (_Float16*)p;
    size_t bytes = halves * 2;
    p += (bytes + 255) & ~(size_t)255;
    return r;
  };
  _Float16* w_other1 = carve(1 * 8 * 512);
  _Float16* w_other2 = carve(4 * 8 * 512);
  _Float16* w_self1  = carve(2 * 8 * 512);
  _Float16* w_self2  = carve(4 * 8 * 512);
  _Float16* w_wu_a   = carve(4 * 256 * 512);
  _Float16* w_wvq_a  = carve(4 * 256 * 512);
  _Float16* w_wvk_a  = carve(4 * 256 * 512);
  _Float16* w_wu_e   = carve(4 * 256 * 512);
  _Float16* w_wvq_e  = carve(4 * 256 * 512);
  _Float16* w_wvk_e  = carve(4 * 256 * 512);
  _Float16* w_ds1    = carve(132 * 8 * 512);
  _Float16* w_ds2    = carve(4 * 9 * 512);
  _Float16* ve_ent   = carve((size_t)B_TOTAL * 128 * 128);
  _Float16* vself    = carve((size_t)B_TOTAL * 128);
  _Float16* catb     = carve((size_t)B_TOTAL * CAT_W);

  auto swz = [&](const float* src, _Float16* dst, int K, int N, int KT, int NT) {
    size_t total = (size_t)KT * NT * 512;
    int blocks = (int)((total + 255) / 256);
    k_swizzle<<<blocks, 256, 0, stream>>>(src, dst, K, N, KT, NT);
  };
  swz(other_W1, w_other1, 32, 128, 1, 8);
  swz(other_W2, w_other2, 128, 128, 4, 8);
  swz(self_W1, w_self1, 48, 128, 2, 8);
  swz(self_W2, w_self2, 128, 128, 4, 8);
  swz(ally_Wu, w_wu_a, 128, 4096, 4, 256);
  swz(ally_Wvq, w_wvq_a, 128, 4096, 4, 256);
  swz(ally_Wvk, w_wvk_a, 128, 4096, 4, 256);
  swz(enemy_Wu, w_wu_e, 128, 4096, 4, 256);
  swz(enemy_Wvq, w_wvq_e, 128, 4096, 4, 256);
  swz(enemy_Wvk, w_wvk_e, 128, 4096, 4, 256);
  swz(ds_W1, w_ds1, 4224, 128, 132, 8);
  swz(ds_W2, w_ds2, 128, 132, 4, 9);

  k_entity_mlp<<<(B_TOTAL * 128) / 64, 256, 0, stream>>>(
      input, w_other1, other_b1, w_other2, other_b2, ve_ent);
  k_self_mlp<<<B_TOTAL / 64, 256, 0, stream>>>(
      input, w_self1, self_b1, w_self2, self_b2, vself, catb);
  k_sampo<<<dim3(B_TOTAL / 16, 2), 256, SAMPO_SMEM, stream>>>(
      ve_ent, vself,
      w_wu_a, ally_bu, w_wvq_a, ally_bvq, w_wvk_a, ally_bvk,
      w_wu_e, enemy_bu, w_wvq_e, enemy_bvq, w_wvk_e, enemy_bvk, catb);
  k_ds_mlp<<<B_TOTAL / 64, 256, 0, stream>>>(
      catb, w_ds1, ds_b1, w_ds2, ds_b2, out);
}
